// PSA_8624294330417
// MI455X (gfx1250) — compile-verified, matrix operand order checked by reading, not run on hardware
//
#include <hip/hip_runtime.h>
#include <hip/hip_bf16.h>

// ---------------------------------------------------------------------------
// PSA block for MI455X (gfx1250, wave32):
//   qkv = Wqkv @ x[:, :256]   (WMMA bf16)
//   flash-attention per (b,h)  (WMMA bf16, f32 online softmax)
//   out = Wproj @ attn         (WMMA bf16)
//   BatchNorm(out) ; concat with skip channels
// LDS tiles laid out so every WMMA fragment loads as 2x ds_load_b128; all
// staging split into batched b128 load phase + store phase; K tiles staged
// with global_load_async_to_lds_b128; packed b128 epilogues.
// ---------------------------------------------------------------------------

typedef __attribute__((ext_vector_type(16))) __bf16 v16bf;
typedef __attribute__((ext_vector_type(8)))  float  v8f;

#define NPIX 4096   // H*W
#define CA   256    // c_attn
#define CIN  512

union B16x8 { uint4 v; __bf16 e8[8]; };
union B16x4 { uint2 v; __bf16 e4[4]; };

// --- fragment builder (CDNA5 ISA 7.12.2 layouts, wave32) --------------------
// 16x32 fragment whose per-lane elements are two contiguous 8-element runs:
// element e -> k = kb + (e<8 ? e : e+8), kb = (lane>=16 ? 8 : 0).
// A fragment from row-major [m][k]; B fragment from TRANSPOSED [n][k].
__device__ __forceinline__ v16bf frag_kcontig(const __bf16* base, int stride, int lane) {
  const int row = lane & 15;
  const int kb  = (lane & 16) ? 8 : 0;
  v16bf f;
#pragma unroll
  for (int e = 0; e < 16; ++e) {
    int k = kb + (e < 8 ? e : e + 8);
    f[e] = base[row * stride + k];
  }
  return f;
}

// --- async global->LDS 16-byte copy (CDNA5 ISA) -------
__device__ __forceinline__ void async_copy_b128(const void* gsrc, void* ldst) {
  unsigned lds_off = (unsigned)(size_t)ldst;   // low 32 bits = LDS byte offset
  asm volatile("global_load_async_to_lds_b128 %0, %1, off"
               :: "v"(lds_off), "v"(gsrc) : "memory");
}
__device__ __forceinline__ void wait_asynccnt0() {
  asm volatile("s_wait_asynccnt 0" ::: "memory");
}

// ===========================================================================
// Kernel 1: qkv = Wqkv(768x256) @ x_attn(256x4096) per batch, bf16 WMMA.
// Output layout: qkv[b][t(q/k/v)][h][n][d]  (bf16, d contiguous)
// grid (32, 6, 4)  block 256 (8 waves); 128x128 tile / block.
// ===========================================================================
#define XT_STRIDE 40   // padded [n][k] stride: 80B, 16B-aligned, bank-skewed

__global__ __launch_bounds__(256) void qkv_gemm_kernel(
    const float* __restrict__ x, const float* __restrict__ w_qkv,
    __bf16* __restrict__ qkv)
{
  __shared__ __bf16 ldsW[128 * 32];          // [o][k] row-major
  __shared__ __bf16 ldsXT[128 * XT_STRIDE];  // [n][k] transposed, padded
  const int t    = threadIdx.x;
  const int lane = t & 31;
  const int wave = t >> 5;
  const int n0 = blockIdx.x * 128;
  const int o0 = blockIdx.y * 128;
  const int b  = blockIdx.z;
  const float* xb = x + (size_t)b * CIN * NPIX;

  v8f acc[8];
#pragma unroll
  for (int j = 0; j < 8; ++j)
#pragma unroll
    for (int r = 0; r < 8; ++r) acc[j][r] = 0.0f;

  for (int kk = 0; kk < 8; ++kk) {
    const int k0 = kk * 32;
    // -- load phase: 8 independent global_load_b128, single wait ------------
    float4 wv[4], xv[4];
#pragma unroll
    for (int i = 0; i < 4; ++i) {            // W tile: 128x32 f32 = 1024 f4
      int e = t + i * 256;
      int row = e >> 3, k4 = e & 7;
      wv[i] = ((const float4*)(w_qkv + (size_t)(o0 + row) * CA + k0))[k4];
    }
#pragma unroll
    for (int i = 0; i < 4; ++i) {            // X tile: 32x128 f32 = 1024 f4
      int e = t + i * 256;
      int k = e >> 5, c4 = (e & 31) * 4;
      xv[i] = *(const float4*)(xb + (size_t)(k0 + k) * NPIX + n0 + c4);
    }
    // -- store phase --------------------------------------------------------
#pragma unroll
    for (int i = 0; i < 4; ++i) {            // W: 4 contiguous bf16 -> b64
      int e = t + i * 256;
      int row = e >> 3, k4 = (e & 7) * 4;
      B16x4 p;
      p.e4[0] = (__bf16)wv[i].x; p.e4[1] = (__bf16)wv[i].y;
      p.e4[2] = (__bf16)wv[i].z; p.e4[3] = (__bf16)wv[i].w;
      *(uint2*)(ldsW + row * 32 + k4) = p.v;
    }
#pragma unroll
    for (int i = 0; i < 4; ++i) {            // X: transposed scatter b16
      int e = t + i * 256;
      int k = e >> 5, c4 = (e & 31) * 4;
      ldsXT[(c4 + 0) * XT_STRIDE + k] = (__bf16)xv[i].x;
      ldsXT[(c4 + 1) * XT_STRIDE + k] = (__bf16)xv[i].y;
      ldsXT[(c4 + 2) * XT_STRIDE + k] = (__bf16)xv[i].z;
      ldsXT[(c4 + 3) * XT_STRIDE + k] = (__bf16)xv[i].w;
    }
    __syncthreads();
    // -- compute: preload all B fragments, then back-to-back WMMA -----------
    v16bf a = frag_kcontig(ldsW + wave * 16 * 32, 32, lane);
    v16bf bfr[8];
#pragma unroll
    for (int j = 0; j < 8; ++j)
      bfr[j] = frag_kcontig(ldsXT + (j * 16) * XT_STRIDE, XT_STRIDE, lane);
#pragma unroll
    for (int j = 0; j < 8; ++j)
      acc[j] = __builtin_amdgcn_wmma_f32_16x16x32_bf16(
          false, a, false, bfr[j], (short)0, acc[j], false, false);
    __syncthreads();
  }

  // packed epilogue: 8 consecutive d per fragment -> one b128 store each
  const int rowoff = (lane & 16) ? 8 : 0;
  const int coll   = lane & 15;
  const int obase  = o0 + wave * 16 + rowoff;   // multiple of 8
  const int tq = obase >> 8;                    // 0=q 1=k 2=v (const over run)
  const int hh = (obase >> 6) & 3;
  const int d0 = obase & 63;
#pragma unroll
  for (int j = 0; j < 8; ++j) {
    const int n = n0 + j * 16 + coll;
    B16x8 pk;
#pragma unroll
    for (int r = 0; r < 8; ++r) pk.e8[r] = (__bf16)acc[j][r];
    *(uint4*)(qkv + ((((size_t)b * 3 + tq) * 4 + hh) * NPIX + n) * 64 + d0) = pk.v;
  }
}

// ===========================================================================
// Kernel 2: flash attention, one block per (b, h, 64-query tile).
// grid (64, 4, 4)  block 128 (4 waves; each wave owns 16 query rows).
// K tile staged async ([m][d]); V tile staged transposed ([d][m]).
// ===========================================================================
#define VT_STRIDE 72   // padded [d][m] stride: 144B, 16B-aligned

__global__ __launch_bounds__(128) void attn_kernel(
    const __bf16* __restrict__ qkv, __bf16* __restrict__ attn_out)
{
  __shared__ __bf16 ldsK[64 * 64];           // [m][d]
  __shared__ __bf16 ldsVT[64 * VT_STRIDE];   // [d][m] transposed, padded
  __shared__ __bf16 ldsP[4 * 16 * 64];       // per-wave P tile [q][m]
  const int t = threadIdx.x, lane = t & 31, wave = t >> 5;
  const int h = blockIdx.y, b = blockIdx.z;
  const int q0 = blockIdx.x * 64;

  const __bf16* qb = qkv + (((size_t)b * 3 + 0) * 4 + h) * (size_t)NPIX * 64;
  const __bf16* kb = qkv + (((size_t)b * 3 + 1) * 4 + h) * (size_t)NPIX * 64;
  const __bf16* vb = qkv + (((size_t)b * 3 + 2) * 4 + h) * (size_t)NPIX * 64;

  const int rowoff = (lane & 16) ? 8 : 0;
  const int coll   = lane & 15;
  const int kbse   = rowoff;                 // fragment k-base

  // Q fragments via 4x global_load_b128, pre-scaled by 1/sqrt(64)=0.125.
  v16bf aq[2];
  {
    const __bf16* qrow = qb + (size_t)(q0 + wave * 16 + coll) * 64;
#pragma unroll
    for (int f = 0; f < 2; ++f) {
      B16x8 r1, r2;
      r1.v = *(const uint4*)(qrow + f * 32 + kbse);
      r2.v = *(const uint4*)(qrow + f * 32 + kbse + 16);
#pragma unroll
      for (int e = 0; e < 8; ++e) {
        aq[f][e]     = (__bf16)(0.125f * (float)r1.e8[e]);
        aq[f][e + 8] = (__bf16)(0.125f * (float)r2.e8[e]);
      }
    }
  }

  float m_i[8], l_i[8];
  v8f oacc[4];
#pragma unroll
  for (int r = 0; r < 8; ++r) { m_i[r] = -3.0e38f; l_i[r] = 0.0f; }
#pragma unroll
  for (int j = 0; j < 4; ++j)
#pragma unroll
    for (int r = 0; r < 8; ++r) oacc[j][r] = 0.0f;

  for (int mt = 0; mt < 64; ++mt) {
    const int m0 = mt * 64;
    if (mt + 1 < 64) {                       // hint next tiles toward L2/WGP$
      __builtin_prefetch(kb + (size_t)(m0 + 64) * 64, 0, 1);
      __builtin_prefetch(vb + (size_t)(m0 + 64) * 64, 0, 1);
    }
    // K tile: 8KB via async global->LDS b128 (4 chunks of 16B per thread)
#pragma unroll
    for (int i = 0; i < 4; ++i) {
      int c = t + i * 128;                   // 512 x 16B chunks
      async_copy_b128(kb + (size_t)m0 * 64 + c * 8, ldsK + c * 8);
    }
    // V tile: batched b128 loads, then transposed scatter [m][d] -> [d][m]
    {
      const uint4* vsrc = (const uint4*)(vb + (size_t)m0 * 64);
      uint4 vv[4];
#pragma unroll
      for (int i = 0; i < 4; ++i) vv[i] = vsrc[t + i * 128];
#pragma unroll
      for (int i = 0; i < 4; ++i) {
        int c = t + i * 128;                 // chunk: 8 bf16, one m, d0..d0+7
        int m = c >> 3, dd0 = (c & 7) * 8;
        B16x8 u; u.v = vv[i];
#pragma unroll
        for (int jj = 0; jj < 8; ++jj)
          ldsVT[(dd0 + jj) * VT_STRIDE + m] = u.e8[jj];
      }
    }
    wait_asynccnt0();
    __syncthreads();

    // S = Q K^T : preload 8 B fragments, then 8 back-to-back WMMAs
    v16bf bk[8];
#pragma unroll
    for (int c = 0; c < 4; ++c)
#pragma unroll
      for (int f = 0; f < 2; ++f)
        bk[c * 2 + f] = frag_kcontig(ldsK + (c * 16) * 64 + f * 32, 64, lane);
    v8f s[4];
#pragma unroll
    for (int c = 0; c < 4; ++c) {
#pragma unroll
      for (int r = 0; r < 8; ++r) s[c][r] = 0.0f;
      s[c] = __builtin_amdgcn_wmma_f32_16x16x32_bf16(
          false, aq[0], false, bk[c * 2 + 0], (short)0, s[c], false, false);
      s[c] = __builtin_amdgcn_wmma_f32_16x16x32_bf16(
          false, aq[1], false, bk[c * 2 + 1], (short)0, s[c], false, false);
    }

    // online softmax: row r of D tile lives in element r across a 16-lane
    // half-wave -> xor-shuffles with mask<16 reduce exactly one row.
    float mnew[8], corr[8];
#pragma unroll
    for (int r = 0; r < 8; ++r) {
      float rm = fmaxf(fmaxf(s[0][r], s[1][r]), fmaxf(s[2][r], s[3][r]));
#pragma unroll
      for (int off = 1; off <= 8; off <<= 1)
        rm = fmaxf(rm, __shfl_xor(rm, off, 32));
      mnew[r] = fmaxf(m_i[r], rm);
      corr[r] = __expf(m_i[r] - mnew[r]);
      m_i[r]  = mnew[r];
    }
#pragma unroll
    for (int c = 0; c < 4; ++c)
#pragma unroll
      for (int r = 0; r < 8; ++r) s[c][r] = __expf(s[c][r] - mnew[r]);
#pragma unroll
    for (int r = 0; r < 8; ++r) {
      float rs = s[0][r] + s[1][r] + s[2][r] + s[3][r];
#pragma unroll
      for (int off = 1; off <= 8; off <<= 1)
        rs += __shfl_xor(rs, off, 32);
      l_i[r] = l_i[r] * corr[r] + rs;
    }
#pragma unroll
    for (int j = 0; j < 4; ++j)
#pragma unroll
      for (int r = 0; r < 8; ++r) oacc[j][r] *= corr[r];

    // P (D-layout) -> per-wave LDS tile [q][m]
    __bf16* pw = ldsP + wave * 16 * 64;
#pragma unroll
    for (int c = 0; c < 4; ++c)
#pragma unroll
      for (int r = 0; r < 8; ++r)
        pw[(rowoff + r) * 64 + c * 16 + coll] = (__bf16)s[c][r];
    asm volatile("s_wait_dscnt 0" ::: "memory");  // same-wave DS store->load

    // O += P V : preload A and all B fragments, then 8 back-to-back WMMAs
    v16bf ap[2];
    ap[0] = frag_kcontig(pw, 64, lane);
    ap[1] = frag_kcontig(pw + 32, 64, lane);
    v16bf bv[8];
#pragma unroll
    for (int j = 0; j < 4; ++j)
#pragma unroll
      for (int f = 0; f < 2; ++f)
        bv[j * 2 + f] =
            frag_kcontig(ldsVT + (j * 16) * VT_STRIDE + f * 32, VT_STRIDE, lane);
#pragma unroll
    for (int j = 0; j < 4; ++j) {
      oacc[j] = __builtin_amdgcn_wmma_f32_16x16x32_bf16(
          false, ap[0], false, bv[j * 2 + 0], (short)0, oacc[j], false, false);
      oacc[j] = __builtin_amdgcn_wmma_f32_16x16x32_bf16(
          false, ap[1], false, bv[j * 2 + 1], (short)0, oacc[j], false, false);
    }
    __syncthreads();
  }

  // packed epilogue: 8 consecutive n per fragment -> one b128 store each
  float rl[8];
#pragma unroll
  for (int r = 0; r < 8; ++r) rl[r] = 1.0f / l_i[r];
  const int nbase = q0 + wave * 16 + rowoff;    // multiple of 8
#pragma unroll
  for (int j = 0; j < 4; ++j) {
    const int c = h * 64 + j * 16 + coll;
    B16x8 pk;
#pragma unroll
    for (int r = 0; r < 8; ++r) pk.e8[r] = (__bf16)(oacc[j][r] * rl[r]);
    *(uint4*)(attn_out + ((size_t)b * CA + c) * NPIX + nbase) = pk.v;
  }
}

// ===========================================================================
// Kernel 3: proj = Wproj(256x256) @ attn_out(256x4096) per batch -> f32 ws.
// grid (32, 2, 4)  block 256.
// ===========================================================================
__global__ __launch_bounds__(256) void proj_gemm_kernel(
    const __bf16* __restrict__ attn_out, const float* __restrict__ w_proj,
    float* __restrict__ proj)
{
  __shared__ __bf16 ldsW[128 * 32];
  __shared__ __bf16 ldsXT[128 * XT_STRIDE];
  const int t    = threadIdx.x;
  const int lane = t & 31;
  const int wave = t >> 5;
  const int n0 = blockIdx.x * 128;
  const int o0 = blockIdx.y * 128;
  const int b  = blockIdx.z;

  v8f acc[8];
#pragma unroll
  for (int j = 0; j < 8; ++j)
#pragma unroll
    for (int r = 0; r < 8; ++r) acc[j][r] = 0.0f;

  for (int kk = 0; kk < 8; ++kk) {
    const int k0 = kk * 32;
    // load phase
    float4 wv[4];
#pragma unroll
    for (int i = 0; i < 4; ++i) {
      int e = t + i * 256;
      int row = e >> 3, k4 = e & 7;
      wv[i] = ((const float4*)(w_proj + (size_t)(o0 + row) * CA + k0))[k4];
    }
    uint4 xv[2];
#pragma unroll
    for (int i = 0; i < 2; ++i) {            // X: bf16, 512 uint4 chunks
      int c = t + i * 256;
      int k = c >> 4, c16 = c & 15;
      xv[i] = ((const uint4*)(attn_out + ((size_t)b * CA + k0 + k) * NPIX + n0))[c16];
    }
    // store phase
#pragma unroll
    for (int i = 0; i < 4; ++i) {
      int e = t + i * 256;
      int row = e >> 3, k4 = (e & 7) * 4;
      B16x4 p;
      p.e4[0] = (__bf16)wv[i].x; p.e4[1] = (__bf16)wv[i].y;
      p.e4[2] = (__bf16)wv[i].z; p.e4[3] = (__bf16)wv[i].w;
      *(uint2*)(ldsW + row * 32 + k4) = p.v;
    }
#pragma unroll
    for (int i = 0; i < 2; ++i) {
      int c = t + i * 256;
      int k = c >> 4, col0 = (c & 15) * 8;
      B16x8 u; u.v = xv[i];
#pragma unroll
      for (int jj = 0; jj < 8; ++jj)
        ldsXT[(col0 + jj) * XT_STRIDE + k] = u.e8[jj];
    }
    __syncthreads();
    v16bf a = frag_kcontig(ldsW + wave * 16 * 32, 32, lane);
    v16bf bfr[8];
#pragma unroll
    for (int j = 0; j < 8; ++j)
      bfr[j] = frag_kcontig(ldsXT + (j * 16) * XT_STRIDE, XT_STRIDE, lane);
#pragma unroll
    for (int j = 0; j < 8; ++j)
      acc[j] = __builtin_amdgcn_wmma_f32_16x16x32_bf16(
          false, a, false, bfr[j], (short)0, acc[j], false, false);
    __syncthreads();
  }

  const int rowoff = (lane & 16) ? 8 : 0;
  const int coll   = lane & 15;
#pragma unroll
  for (int j = 0; j < 8; ++j) {
    const int n = n0 + j * 16 + coll;
#pragma unroll
    for (int r = 0; r < 8; ++r) {
      const int o = o0 + wave * 16 + rowoff + r;
      proj[((size_t)b * CA + o) * NPIX + n] = acc[j][r];
    }
  }
}

// ===========================================================================
// Kernel 4: per-channel BN stats over (B, N) -> stats[c]=mean, stats[256+c]=rstd
// ===========================================================================
__global__ __launch_bounds__(256) void bn_stats_kernel(
    const float* __restrict__ proj, float* __restrict__ stats)
{
  __shared__ float ss[256], sq[256];
  const int c = blockIdx.x, t = threadIdx.x;
  float a = 0.0f, a2 = 0.0f;
  for (int i = t; i < 4096; i += 256) {        // 4096 float4 per channel
    const int b = i >> 10, n4 = i & 1023;
    const float4 v = *(const float4*)(proj + ((size_t)b * CA + c) * NPIX + n4 * 4);
    a  += v.x + v.y + v.z + v.w;
    a2 += v.x * v.x + v.y * v.y + v.z * v.z + v.w * v.w;
  }
  ss[t] = a; sq[t] = a2;
  __syncthreads();
  for (int s = 128; s > 0; s >>= 1) {
    if (t < s) { ss[t] += ss[t + s]; sq[t] += sq[t + s]; }
    __syncthreads();
  }
  if (t == 0) {
    const float inv  = 1.0f / (4.0f * NPIX);
    const float mean = ss[0] * inv;
    const float var  = sq[0] * inv - mean * mean;
    stats[c]      = mean;
    stats[CA + c] = rsqrtf(var + 1e-5f);
  }
}

// ===========================================================================
// Kernel 5: normalize + gamma/beta for channels <256, skip-copy for >=256.
// ===========================================================================
__global__ __launch_bounds__(256) void finalize_kernel(
    const float* __restrict__ proj, const float* __restrict__ x,
    const float* __restrict__ stats, const float* __restrict__ gamma,
    const float* __restrict__ beta, float* __restrict__ out)
{
  const size_t idx4 = (size_t)blockIdx.x * 256 + threadIdx.x;
  const size_t flat = idx4 * 4;                 // element index in (B,512,4096)
  const int b   = (int)(flat >> 21);            // 512*4096 = 2^21
  const int rem = (int)(flat & ((1u << 21) - 1));
  const int c   = rem >> 12;
  const int n   = rem & 4095;
  float4* o4 = (float4*)out;
  if (c < CA) {
    const size_t pidx = (((size_t)b * CA + c) * NPIX + n) >> 2;
    const float4 v = ((const float4*)proj)[pidx];
    const float g  = gamma[c] * stats[CA + c];
    const float bb = beta[c] - stats[c] * g;    // (v-mean)*rstd*g+beta = v*g+bb
    float4 r;
    r.x = v.x * g + bb; r.y = v.y * g + bb;
    r.z = v.z * g + bb; r.w = v.w * g + bb;
    o4[idx4] = r;
  } else {
    o4[idx4] = ((const float4*)x)[idx4];
  }
}

// ===========================================================================
extern "C" void kernel_launch(void* const* d_in, const int* in_sizes, int n_in,
                              void* d_out, int out_size, void* d_ws, size_t ws_size,
                              hipStream_t stream) {
  const float* x      = (const float*)d_in[0];
  const float* w_qkv  = (const float*)d_in[1];
  const float* w_proj = (const float*)d_in[2];
  const float* gamma  = (const float*)d_in[3];
  const float* beta   = (const float*)d_in[4];
  float* out = (float*)d_out;

  // workspace layout
  __bf16* qkv  = (__bf16*)d_ws;                        // 4*3*4*4096*64 bf16
  __bf16* attn = qkv + (size_t)4 * 3 * 4 * NPIX * 64;  // 4*256*4096 bf16
  float*  proj = (float*)(attn + (size_t)4 * CA * NPIX);
  float*  stats = proj + (size_t)4 * CA * NPIX;        // 512 f32

  qkv_gemm_kernel<<<dim3(32, 6, 4), 256, 0, stream>>>(x, w_qkv, qkv);
  attn_kernel<<<dim3(64, 4, 4), 128, 0, stream>>>(qkv, attn);
  proj_gemm_kernel<<<dim3(32, 2, 4), 256, 0, stream>>>(attn, w_proj, proj);
  bn_stats_kernel<<<256, 256, 0, stream>>>(proj, stats);
  finalize_kernel<<<8192, 256, 0, stream>>>(proj, x, stats, gamma, beta, out);
}
